// PooledTransformerDecoder_36601711297062
// MI455X (gfx1250) — matrix operand order repaired and checked by reading, not compiled, not run on hardware
//
#include <hip/hip_runtime.h>
#include <hip/hip_bf16.h>
#include <math.h>

#define DMODEL 512
#define NHEAD  8
#define HDIM   64
#define SEQ    64
#define FFDIM  2048
#define NBATCH 32
#define NVOCAB 32000
#define NLAYERS 4

typedef __bf16 bf16;
typedef __attribute__((ext_vector_type(16))) __bf16 v16bf;
typedef __attribute__((ext_vector_type(8)))  float  v8f;
typedef unsigned int u32x4 __attribute__((ext_vector_type(4)));
typedef int          i32x4 __attribute__((ext_vector_type(4)));
typedef int          i32x8 __attribute__((ext_vector_type(8)));

// ---- CDNA5 feature detection (fallback chain: TDM -> async-LDS -> manual) ----
#if defined(__has_builtin)
# if __has_builtin(__builtin_amdgcn_tensor_load_to_lds)
#  define HAVE_TDM 1
# endif
# if __has_builtin(__builtin_amdgcn_global_load_async_to_lds_b128)
#  define HAVE_ASYNC 1
# endif
# if __has_builtin(__builtin_amdgcn_s_wait_tensorcnt)
#  define TDM_WAIT() __builtin_amdgcn_s_wait_tensorcnt((short)0)
# endif
# if __has_builtin(__builtin_amdgcn_s_wait_asynccnt)
#  define ASYNC_WAIT() __builtin_amdgcn_s_wait_asynccnt((short)0)
# endif
#endif
#ifndef TDM_WAIT
# define TDM_WAIT() asm volatile("s_wait_tensorcnt 0x0" ::: "memory")
#endif
#ifndef ASYNC_WAIT
# define ASYNC_WAIT() asm volatile("s_wait_asynccnt 0x0" ::: "memory")
#endif

static __device__ __forceinline__ bf16 f2bf(float f) {
  unsigned u = __builtin_bit_cast(unsigned, f);
  u += 0x7fffu + ((u >> 16) & 1u);
  unsigned short h = (unsigned short)(u >> 16);
  return __builtin_bit_cast(bf16, h);
}

// Load a 16-element bf16 fragment as two 16-byte chunks: p[0..7] and p[off2..off2+7].
// A-fragment (16x32 bf16): lane holds 16 contiguous k (base 0 or 16)          -> off2 = 8.
// B-fragment (32x16 bf16): lane holds 16 contiguous rows (base 0 or 16)        -> off2 = 8.
static __device__ __forceinline__ v16bf load_frag(const bf16* p, int off2) {
  v16bf r;
  *(uint4*)&r = *(const uint4*)p;
  *(((uint4*)&r) + 1) = *(const uint4*)(p + off2);
  return r;
}

#if defined(HAVE_TDM)
// Issue a 2D TDM load: tile (elems_x x rows, 2-byte elements) from global
// (row stride = stride_elems) into LDS at lds_off. D# packed per ISA 8.3/8.4:
//  group0: count=1 | lds_addr | global_addr[56:0] | type=2
//  group1: data_size=2B | tensor_dim0/1 | tile_dim0/1 | tensor_dim0_stride
// Rows beyond tensor_h read as zero (TDM OOB semantics) -> implicit M-padding.
static __device__ __forceinline__ void tdm_load_2d_bf16(
    const void* gaddr, unsigned lds_off, int elems_x, int rows,
    int tensor_w, int tensor_h, long long stride_elems)
{
  const unsigned long long ga = (unsigned long long)(size_t)gaddr;
  u32x4 g0;
  g0.x = 1u;                                                  // count=1 (valid, user)
  g0.y = lds_off;                                             // lds_addr (bytes)
  g0.z = (unsigned)(ga & 0xffffffffu);                        // global_addr[31:0]
  g0.w = (unsigned)((ga >> 32) & 0x01ffffffu) | 0x80000000u;  // addr[56:32] | type=2
  i32x8 g1;
  g1[0] = 1 << 16;                                            // data_size = 1 -> 2 bytes
  g1[1] = (int)(((unsigned)tensor_w & 0xffffu) << 16);        // tensor_dim0[15:0]
  g1[2] = (int)((((unsigned)tensor_w >> 16) & 0xffffu) |
                (((unsigned)tensor_h & 0xffffu) << 16));      // td0[31:16] | td1[15:0]
  g1[3] = (int)((((unsigned)tensor_h >> 16) & 0xffffu) |
                (((unsigned)elems_x & 0xffffu) << 16));       // td1[31:16] | tile_dim0
  g1[4] = (int)((unsigned)rows & 0xffffu);                    // tile_dim1 (tile_dim2=0)
  g1[5] = (int)(unsigned)((unsigned long long)stride_elems & 0xffffffffu); // td0_stride[31:0]
  g1[6] = (int)(unsigned)(((unsigned long long)stride_elems >> 32) & 0xffffu); // [47:32]
  g1[7] = 0;
  const i32x4 z4 = {0, 0, 0, 0};
#if __clang_major__ >= 23
  const i32x8 z8 = {0, 0, 0, 0, 0, 0, 0, 0};
  __builtin_amdgcn_tensor_load_to_lds(g0, g1, z4, z4, z8, 0);
#else
  __builtin_amdgcn_tensor_load_to_lds(g0, g1, z4, z4, 0);
#endif
}
#endif

// ---------------------------------------------------------------------------
// Generic tiled WMMA GEMM: C[M,N] (f32, row stride ldc) = A[M,K]bf16 @ B[K,N]bf16 + bias
// Optional relu epilogue and bf16 shadow output Cb (same ldc).
// Block: 256 threads = 8 waves (4x2), wave computes 32x32 via 2x2 WMMA tiles, BK=32.
// A tile (128x32, row stride K) staged by the Tensor Data Mover when available.
// ---------------------------------------------------------------------------
#define BM 128
#define BN 64
#define BK 32

__global__ __launch_bounds__(256, 2)
void k_gemm_bf16(const bf16* __restrict__ A, const bf16* __restrict__ Bm,
                 const float* __restrict__ bias, float* __restrict__ C,
                 bf16* __restrict__ Cb, int M, int N, int K,
                 long long ldc, int relu)
{
  __shared__ __attribute__((aligned(16))) bf16 As[BM][BK];   // 8 KB
  __shared__ __attribute__((aligned(16))) bf16 Bs[BN][BK];   // 4 KB, stored [n][k]
  const int tid  = threadIdx.x;
  const int lane = tid & 31, wave = tid >> 5;
  const int wm = wave >> 1, wn = wave & 1;
  const int bm0 = blockIdx.y * BM, bn0 = blockIdx.x * BN;
  const int am    = lane & 15;
  const int koffA = (lane & 16) ? 16 : 0;
  const int bc    = lane & 15;
  const int roffB = (lane & 16) ? 16 : 0;
  v8f acc[2][2] = {};

  const int ar = tid >> 1;            // A staging row (manual path)
  const int ac = (tid & 1) << 4;      // 0 or 16
  const int bn = tid & 63;            // B staging column (n)
  const int bk = (tid >> 6) << 3;     // 8-wide k group

  for (int k0 = 0; k0 < K; k0 += BK) {
#if defined(HAVE_TDM)
    if (wave == 0) {                  // one 8KB DMA per block per K-step
      tdm_load_2d_bf16(A + (size_t)bm0 * K + k0,
                       (unsigned)(size_t)(void*)&As[0][0],
                       BK, BM, K, M - bm0, (long long)K);
      TDM_WAIT();
    }
#elif defined(HAVE_ASYNC)
    {
      const int gr = bm0 + ar;
      if (gr < M) {
        const bf16* src = A + (size_t)gr * K + (k0 + ac);
        __builtin_amdgcn_global_load_async_to_lds_b128(
            (const __attribute__((address_space(1))) void*)src,
            (__attribute__((address_space(3))) void*)&As[ar][ac], 0, 0);
        __builtin_amdgcn_global_load_async_to_lds_b128(
            (const __attribute__((address_space(1))) void*)(src + 8),
            (__attribute__((address_space(3))) void*)&As[ar][ac + 8], 0, 0);
      } else {
        for (int i = 0; i < 16; ++i) As[ar][ac + i] = f2bf(0.f);
      }
      ASYNC_WAIT();
    }
#else
    { // manual stage of A tile (row-major, k-contiguous)
      const int gr = bm0 + ar;
      bf16* dst = &As[ar][ac];
      if (gr < M) {
        const bf16* src = A + (size_t)gr * K + (k0 + ac);
        *(uint4*)dst       = *(const uint4*)src;
        *(uint4*)(dst + 8) = *(const uint4*)(src + 8);
        if (k0 + BK < K) __builtin_prefetch(src + BK, 0, 0);
      } else {
        for (int i = 0; i < 16; ++i) dst[i] = f2bf(0.f);
      }
    }
#endif
    { // stage B tile transposed to [n][k] so B-fragments are contiguous loads
      const int gn = bn0 + bn;
      if (gn < N) {
        const bf16* src = &Bm[(size_t)(k0 + bk) * N + gn];
        for (int i = 0; i < 8; ++i)
          Bs[bn][bk + i] = src[(size_t)i * N];
        if (k0 + BK < K) __builtin_prefetch(src + (size_t)BK * N, 0, 0);
      } else {
        for (int i = 0; i < 8; ++i) Bs[bn][bk + i] = f2bf(0.f);
      }
    }
    __syncthreads();
    v16bf afr[2], bfr[2];
    for (int i = 0; i < 2; ++i)
      afr[i] = load_frag(&As[wm * 32 + i * 16 + am][koffA], 8);
    for (int j = 0; j < 2; ++j)
      bfr[j] = load_frag(&Bs[wn * 32 + j * 16 + bc][roffB], 8);
    for (int i = 0; i < 2; ++i)
      for (int j = 0; j < 2; ++j)
        acc[i][j] = __builtin_amdgcn_wmma_f32_16x16x32_bf16(
            false, afr[i], false, bfr[j], (short)0, acc[i][j], false, false);
    __syncthreads();
  }

  const int rofC = (lane & 16) ? 8 : 0;
  for (int j = 0; j < 2; ++j) {
    const int gcol = bn0 + wn * 32 + j * 16 + bc;
    if (gcol >= N) continue;
    const float bv = bias ? bias[gcol] : 0.f;
    for (int i = 0; i < 2; ++i) {
      for (int v = 0; v < 8; ++v) {
        const int grow = bm0 + wm * 32 + i * 16 + v + rofC;
        if (grow < M) {
          float val = acc[i][j][v] + bv;
          if (relu) val = fmaxf(val, 0.f);
          const size_t o = (size_t)grow * (size_t)ldc + gcol;
          C[o] = val;
          if (Cb) Cb[o] = f2bf(val);
        }
      }
    }
  }
}

// ---------------------------------------------------------------------------
// Fused self-attention for one (batch, head): scores = (Q/8)K^T, masked softmax
// over keys < L, out = S @ V. All matmuls via WMMA; Q/K/V staged f32->bf16 in LDS.
// Block: 128 threads (4 waves), wave handles 16 query rows. EXEC all-ones at WMMA.
// ---------------------------------------------------------------------------
__global__ __launch_bounds__(128, 2)
void k_attn(const float* __restrict__ Q, const float* __restrict__ Kv,
            const float* __restrict__ V, bf16* __restrict__ Ob, int L)
{
  __shared__ __attribute__((aligned(16))) bf16 Qs[SEQ][HDIM];   // [q][hd]
  __shared__ __attribute__((aligned(16))) bf16 Ks[SEQ][HDIM];   // [key][hd]
  __shared__ __attribute__((aligned(16))) bf16 Vt[HDIM][SEQ];   // [hd][key] (transposed)
  __shared__ __attribute__((aligned(16))) bf16 Ss[SEQ][SEQ];    // probs [q][key]
  const int b = blockIdx.x >> 3;
  const int h = blockIdx.x & 7;
  const size_t base = ((size_t)b * SEQ) * DMODEL + (size_t)h * HDIM;
  const int tid = threadIdx.x, lane = tid & 31, wave = tid >> 5;
  const int Lp = (L + 15) & ~15;

  for (int idx = tid; idx < SEQ * HDIM; idx += 128) {
    const int r = idx >> 6, c = idx & 63;
    const size_t g = base + (size_t)r * DMODEL + c;
    const bool vld = (r < L);
    Qs[r][c] = f2bf(vld ? Q[g] * 0.125f : 0.f);   // fold 1/sqrt(64) into Q
    Ks[r][c] = f2bf(vld ? Kv[g] : 0.f);
    Vt[c][r] = f2bf(vld ? V[g] : 0.f);
    ((bf16*)Ss)[idx] = f2bf(0.f);                 // zero 64x64 prob matrix
  }
  __syncthreads();

  const int q0 = wave << 4;
  const int am = lane & 15, koffA = (lane & 16) ? 16 : 0;
  const int bc = lane & 15, roffB = (lane & 16) ? 16 : 0;
  const int rofC = (lane & 16) ? 8 : 0;
  const int ntiles = Lp >> 4;

  if (q0 < Lp) {  // wave-uniform
    v8f sc[4] = {};
    for (int kk = 0; kk < HDIM; kk += 32) {
      const v16bf a = load_frag(&Qs[q0 + am][kk + koffA], 8);
      for (int n = 0; n < ntiles; ++n) {
        const v16bf bb = load_frag(&Ks[(n << 4) + bc][kk + roffB], 8);
        sc[n] = __builtin_amdgcn_wmma_f32_16x16x32_bf16(
            false, a, false, bb, (short)0, sc[n], false, false);
      }
    }
    // row-wise softmax: C layout -> row = v + (lane>=16 ? 8 : 0), col = n*16 + lane%16.
    for (int v = 0; v < 8; ++v) {
      float sv[4];
      float mx = -1e30f;
      for (int n = 0; n < ntiles; ++n) {
        const int col = (n << 4) + bc;
        const float s = (col < L) ? sc[n][v] : -1e30f;   // mask padded keys
        sv[n] = s;
        mx = fmaxf(mx, s);
      }
      for (int off = 1; off < 16; off <<= 1) mx = fmaxf(mx, __shfl_xor(mx, off));
      float sum = 0.f;
      for (int n = 0; n < ntiles; ++n) { sv[n] = __expf(sv[n] - mx); sum += sv[n]; }
      for (int off = 1; off < 16; off <<= 1) sum += __shfl_xor(sum, off);
      const float inv = 1.f / sum;
      const int row = q0 + v + rofC;
      for (int n = 0; n < ntiles; ++n) Ss[row][(n << 4) + bc] = f2bf(sv[n] * inv);
    }
  }
  __syncthreads();

  if (q0 < Lp) {
    v8f oc[4] = {};
    for (int kk = 0; kk < Lp; kk += 32) {     // padded cols of Ss are zero
      const v16bf a = load_frag(&Ss[q0 + am][kk + koffA], 8);
      for (int n = 0; n < 4; ++n) {
        const v16bf bb = load_frag(&Vt[(n << 4) + bc][kk + roffB], 8);
        oc[n] = __builtin_amdgcn_wmma_f32_16x16x32_bf16(
            false, a, false, bb, (short)0, oc[n], false, false);
      }
    }
    for (int n = 0; n < 4; ++n) {
      const int col = (n << 4) + bc;
      for (int v = 0; v < 8; ++v) {
        const int row = q0 + v + rofC;
        if (row < L) Ob[base + (size_t)row * DMODEL + col] = f2bf(oc[n][v]);
      }
    }
  }
}

// ---------------------------------------------------------------------------
// y = LayerNorm(x + r) * g + b ; writes f32 (y) and bf16 (yb), may alias x.
// r_per_batch: residual is a per-batch D-vector (collapsed cross-attention).
// ---------------------------------------------------------------------------
__global__ __launch_bounds__(128)
void k_add_ln(const float* __restrict__ x, const float* __restrict__ r, int r_per_batch,
              const float* __restrict__ g, const float* __restrict__ bt,
              float* __restrict__ y, bf16* __restrict__ yb)
{
  const int row = blockIdx.x;           // b*SEQ + pos
  const int tid = threadIdx.x;          // 128 threads x 4 elems
  const size_t xo = (size_t)row * DMODEL;
  const size_t ro = r_per_batch ? (size_t)(row >> 6) * DMODEL : xo;
  float vals[4];
  float s = 0.f, s2 = 0.f;
  for (int i = 0; i < 4; ++i) {
    const int c = tid * 4 + i;
    const float vv = x[xo + c] + r[ro + c];
    vals[i] = vv; s += vv; s2 += vv * vv;
  }
  for (int off = 1; off < 32; off <<= 1) { s += __shfl_xor(s, off); s2 += __shfl_xor(s2, off); }
  __shared__ float rs1[4];
  __shared__ float rs2[4];
  const int wv = tid >> 5, ln = tid & 31;
  if (ln == 0) { rs1[wv] = s; rs2[wv] = s2; }
  __syncthreads();
  s  = rs1[0] + rs1[1] + rs1[2] + rs1[3];
  s2 = rs2[0] + rs2[1] + rs2[2] + rs2[3];
  const float mu  = s * (1.f / DMODEL);
  const float var = s2 * (1.f / DMODEL) - mu * mu;
  const float rcs = rsqrtf(var + 1e-5f);
  for (int i = 0; i < 4; ++i) {
    const int c = tid * 4 + i;
    const float o = (vals[i] - mu) * rcs * g[c] + bt[c];
    y[xo + c]  = o;
    yb[xo + c] = f2bf(o);
  }
}

// Cross-attention collapses: identical keys -> uniform softmax -> out = V.
// ca_const[l][b] = (memory[b] @ ca_wv[l] + ca_bv[l]) @ ca_wo[l] + ca_bo[l]
__global__ __launch_bounds__(256)
void k_ca_const(const float* __restrict__ memory, const float* __restrict__ ca_wv,
                const float* __restrict__ ca_bv, const float* __restrict__ ca_wo,
                const float* __restrict__ ca_bo, float* __restrict__ out)
{
  const int l = blockIdx.x >> 5;
  const int b = blockIdx.x & 31;
  __shared__ float vsh[DMODEL];
  const float* mrow = memory + (size_t)b * DMODEL;
  const float* wv = ca_wv + (size_t)l * DMODEL * DMODEL;
  const float* wo = ca_wo + (size_t)l * DMODEL * DMODEL;
  const int tid = threadIdx.x;
  for (int j = tid; j < DMODEL; j += 256) {
    float s = ca_bv[l * DMODEL + j];
    for (int k = 0; k < DMODEL; ++k) s += mrow[k] * wv[(size_t)k * DMODEL + j];
    vsh[j] = s;
  }
  __syncthreads();
  for (int j = tid; j < DMODEL; j += 256) {
    float s = ca_bo[l * DMODEL + j];
    for (int k = 0; k < DMODEL; ++k) s += vsh[k] * wo[(size_t)k * DMODEL + j];
    out[((size_t)l * NBATCH + b) * DMODEL + j] = s;
  }
}

__global__ __launch_bounds__(256)
void k_embed_step(const float* __restrict__ emb, const int* __restrict__ tokens,
                  float* __restrict__ xf, bf16* __restrict__ xb, int t)
{
  const int b = blockIdx.x;
  const int tok = (t == 0) ? 0 : tokens[b * (SEQ + 1) + t];
  const size_t o = ((size_t)b * SEQ + t) * DMODEL;
  for (int c = threadIdx.x; c < DMODEL; c += 256) {
    const int i2 = c & ~1;
    const float ang = (float)t * __expf((float)i2 * (-9.210340371976184f / (float)DMODEL));
    const float pe  = (c & 1) ? __cosf(ang) : __sinf(ang);
    const float val = emb[(size_t)tok * DMODEL + c] + pe;
    xf[o + c] = val;
    xb[o + c] = f2bf(val);
  }
}

__global__ __launch_bounds__(256)
void k_cvt_bf16(const float* __restrict__ s, bf16* __restrict__ d, long long n) {
  long long i = (long long)blockIdx.x * 256 + threadIdx.x;
  const long long stride = (long long)gridDim.x * 256;
  for (; i < n; i += stride) d[i] = f2bf(s[i]);
}

__global__ __launch_bounds__(256)
void k_zero_u32(unsigned* __restrict__ p, long long n) {
  long long i = (long long)blockIdx.x * 256 + threadIdx.x;
  const long long stride = (long long)gridDim.x * 256;
  for (; i < n; i += stride) p[i] = 0u;
}

__global__ __launch_bounds__(256)
void k_gather_last(const bf16* __restrict__ xb, bf16* __restrict__ lastb, int t) {
  const int b = blockIdx.x;
  for (int c = threadIdx.x; c < DMODEL; c += 256)
    lastb[b * DMODEL + c] = xb[((size_t)b * SEQ + t) * DMODEL + c];
}

// jnp.argmax semantics: first index of the max.
__global__ __launch_bounds__(256)
void k_argmax(const float* __restrict__ logits, int t, int* __restrict__ tokens,
              float* __restrict__ gen)
{
  const int b = blockIdx.x;
  const float* row = logits + (size_t)b * SEQ * NVOCAB + (size_t)t * NVOCAB;
  __shared__ float sv[256];
  __shared__ int   si[256];
  float best = -3.4e38f; int bi = 0;
  for (int i = threadIdx.x; i < NVOCAB; i += 256) {
    const float v = row[i];
    if (v > best) { best = v; bi = i; }   // ascending i -> first max per thread
  }
  sv[threadIdx.x] = best; si[threadIdx.x] = bi;
  __syncthreads();
  for (int s = 128; s > 0; s >>= 1) {
    if (threadIdx.x < s) {
      const float v2 = sv[threadIdx.x + s]; const int i2 = si[threadIdx.x + s];
      if (v2 > sv[threadIdx.x] || (v2 == sv[threadIdx.x] && i2 < si[threadIdx.x])) {
        sv[threadIdx.x] = v2; si[threadIdx.x] = i2;
      }
    }
    __syncthreads();
  }
  if (threadIdx.x == 0) {
    tokens[b * (SEQ + 1) + (t + 1)] = si[0];
    gen[b * SEQ + t] = (float)si[0];
  }
}

// ---------------------------------------------------------------------------
extern "C" void kernel_launch(void* const* d_in, const int* in_sizes, int n_in,
                              void* d_out, int out_size, void* d_ws, size_t ws_size,
                              hipStream_t stream)
{
  (void)in_sizes; (void)n_in; (void)out_size; (void)ws_size;
  const float* memory    = (const float*)d_in[0];
  const float* embedding = (const float*)d_in[1];
  const float* sa_wq = (const float*)d_in[2];
  const float* sa_wk = (const float*)d_in[3];
  const float* sa_wv = (const float*)d_in[4];
  const float* sa_wo = (const float*)d_in[5];
  const float* sa_bq = (const float*)d_in[6];
  const float* sa_bk = (const float*)d_in[7];
  const float* sa_bv = (const float*)d_in[8];
  const float* sa_bo = (const float*)d_in[9];
  // ca_wq (10) / ca_wk (11) / ca_bq (14) / ca_bk (15) unused: cross-attn collapses.
  const float* ca_wv = (const float*)d_in[12];
  const float* ca_wo = (const float*)d_in[13];
  const float* ca_bv = (const float*)d_in[16];
  const float* ca_bo = (const float*)d_in[17];
  const float* ff_w1 = (const float*)d_in[18];
  const float* ff_b1 = (const float*)d_in[19];
  const float* ff_w2 = (const float*)d_in[20];
  const float* ff_b2 = (const float*)d_in[21];
  const float* ln1_g = (const float*)d_in[22];
  const float* ln1_b = (const float*)d_in[23];
  const float* ln2_g = (const float*)d_in[24];
  const float* ln2_b = (const float*)d_in[25];
  const float* ln3_g = (const float*)d_in[26];
  const float* ln3_b = (const float*)d_in[27];
  const float* fc_w  = (const float*)d_in[28];
  const float* fc_b  = (const float*)d_in[29];

  // --- workspace carve (256B aligned) ---
  char* wp = (char*)d_ws;
  auto carve = [&](size_t bytes) { void* r = (void*)wp; wp += (bytes + 255) & ~(size_t)255; return r; };
  bf16*  wqB   = (bf16*)carve((size_t)NLAYERS * DMODEL * DMODEL * 2);
  bf16*  wkB   = (bf16*)carve((size_t)NLAYERS * DMODEL * DMODEL * 2);
  bf16*  wvB   = (bf16*)carve((size_t)NLAYERS * DMODEL * DMODEL * 2);
  bf16*  woB   = (bf16*)carve((size_t)NLAYERS * DMODEL * DMODEL * 2);
  bf16*  w1B   = (bf16*)carve((size_t)NLAYERS * DMODEL * FFDIM * 2);
  bf16*  w2B   = (bf16*)carve((size_t)NLAYERS * FFDIM * DMODEL * 2);
  bf16*  fcwB  = (bf16*)carve((size_t)DMODEL * NVOCAB * 2);
  float* caC   = (float*)carve((size_t)NLAYERS * NBATCH * DMODEL * 4);
  float* embF  = (float*)carve((size_t)NBATCH * SEQ * DMODEL * 4);
  bf16*  embB  = (bf16*)carve((size_t)NBATCH * SEQ * DMODEL * 2);
  float* xF    = (float*)carve((size_t)NBATCH * SEQ * DMODEL * 4);
  bf16*  xB    = (bf16*)carve((size_t)NBATCH * SEQ * DMODEL * 2);
  float* qF    = (float*)carve((size_t)NBATCH * SEQ * DMODEL * 4);
  float* kF    = (float*)carve((size_t)NBATCH * SEQ * DMODEL * 4);
  float* vF    = (float*)carve((size_t)NBATCH * SEQ * DMODEL * 4);
  bf16*  attnB = (bf16*)carve((size_t)NBATCH * SEQ * DMODEL * 2);
  float* t1F   = (float*)carve((size_t)NBATCH * SEQ * DMODEL * 4);
  float* hF    = (float*)carve((size_t)NBATCH * SEQ * FFDIM * 4);
  bf16*  hB    = (bf16*)carve((size_t)NBATCH * SEQ * FFDIM * 2);
  bf16*  lastB = (bf16*)carve((size_t)NBATCH * DMODEL * 2);
  int*   tok   = (int*)carve((size_t)NBATCH * (SEQ + 1) * 4);

  // --- setup: weight bf16 conversion, buffers, collapsed cross-attn ---
  k_cvt_bf16<<<1024, 256, 0, stream>>>(sa_wq, wqB, (long long)NLAYERS * DMODEL * DMODEL);
  k_cvt_bf16<<<1024, 256, 0, stream>>>(sa_wk, wkB, (long long)NLAYERS * DMODEL * DMODEL);
  k_cvt_bf16<<<1024, 256, 0, stream>>>(sa_wv, wvB, (long long)NLAYERS * DMODEL * DMODEL);
  k_cvt_bf16<<<1024, 256, 0, stream>>>(sa_wo, woB, (long long)NLAYERS * DMODEL * DMODEL);
  k_cvt_bf16<<<2048, 256, 0, stream>>>(ff_w1, w1B, (long long)NLAYERS * DMODEL * FFDIM);
  k_cvt_bf16<<<2048, 256, 0, stream>>>(ff_w2, w2B, (long long)NLAYERS * FFDIM * DMODEL);
  k_cvt_bf16<<<4096, 256, 0, stream>>>(fc_w,  fcwB, (long long)DMODEL * NVOCAB);
  k_zero_u32<<<1024, 256, 0, stream>>>((unsigned*)embF, (long long)NBATCH * SEQ * DMODEL);
  k_zero_u32<<<512,  256, 0, stream>>>((unsigned*)embB, (long long)NBATCH * SEQ * DMODEL / 2);
  k_zero_u32<<<16,   256, 0, stream>>>((unsigned*)tok,  (long long)NBATCH * (SEQ + 1));
  k_ca_const<<<NLAYERS * NBATCH, 256, 0, stream>>>(memory, ca_wv, ca_bv, ca_wo, ca_bo, caC);

  const int M = NBATCH * SEQ;  // fixed padded rows: 2048
  dim3 blk(256);
  dim3 gD((DMODEL + BN - 1) / BN, (M + BM - 1) / BM);   // (8,16)
  dim3 gF((FFDIM + BN - 1) / BN, (M + BM - 1) / BM);    // (32,16)
  dim3 gL((NVOCAB + BN - 1) / BN, 1);                   // (500,1)
  float* gen = (float*)d_out;
  float* logitsBase = (float*)d_out + NBATCH * SEQ;     // logits region (B,64,V)

  for (int t = 0; t < SEQ; ++t) {
    const int L = t + 1;
    k_embed_step<<<NBATCH, 256, 0, stream>>>(embedding, tok, embF, embB, t);
    hipMemcpyAsync(xF, embF, (size_t)M * DMODEL * sizeof(float), hipMemcpyDeviceToDevice, stream);
    hipMemcpyAsync(xB, embB, (size_t)M * DMODEL * sizeof(bf16),  hipMemcpyDeviceToDevice, stream);

    for (int l = 0; l < NLAYERS; ++l) {
      const bf16* wq = wqB + (size_t)l * DMODEL * DMODEL;
      const bf16* wk = wkB + (size_t)l * DMODEL * DMODEL;
      const bf16* wv = wvB + (size_t)l * DMODEL * DMODEL;
      const bf16* wo = woB + (size_t)l * DMODEL * DMODEL;
      const bf16* w1 = w1B + (size_t)l * DMODEL * FFDIM;
      const bf16* w2 = w2B + (size_t)l * FFDIM * DMODEL;

      // self-attention
      k_gemm_bf16<<<gD, blk, 0, stream>>>(xB, wq, sa_bq + l * DMODEL, qF, nullptr, M, DMODEL, DMODEL, DMODEL, 0);
      k_gemm_bf16<<<gD, blk, 0, stream>>>(xB, wk, sa_bk + l * DMODEL, kF, nullptr, M, DMODEL, DMODEL, DMODEL, 0);
      k_gemm_bf16<<<gD, blk, 0, stream>>>(xB, wv, sa_bv + l * DMODEL, vF, nullptr, M, DMODEL, DMODEL, DMODEL, 0);
      k_attn<<<NBATCH * NHEAD, 128, 0, stream>>>(qF, kF, vF, attnB, L);
      k_gemm_bf16<<<gD, blk, 0, stream>>>(attnB, wo, sa_bo + l * DMODEL, t1F, nullptr, M, DMODEL, DMODEL, DMODEL, 0);
      k_add_ln<<<M, 128, 0, stream>>>(xF, t1F, 0, ln1_g + l * DMODEL, ln1_b + l * DMODEL, xF, xB);

      // cross-attention (collapsed to per-batch constant) + LN2
      k_add_ln<<<M, 128, 0, stream>>>(xF, caC + (size_t)l * NBATCH * DMODEL, 1,
                                      ln2_g + l * DMODEL, ln2_b + l * DMODEL, xF, xB);

      // FFN + LN3
      k_gemm_bf16<<<gF, blk, 0, stream>>>(xB, w1, ff_b1 + l * FFDIM, hF, hB, M, FFDIM, DMODEL, FFDIM, 1);
      k_gemm_bf16<<<gD, blk, 0, stream>>>(hB, w2, ff_b2 + l * DMODEL, t1F, nullptr, M, DMODEL, FFDIM, DMODEL, 0);
      k_add_ln<<<M, 128, 0, stream>>>(xF, t1F, 0, ln3_g + l * DMODEL, ln3_b + l * DMODEL, xF, xB);
    }

    // logits = x[:, t, :] @ fc_w + fc_b  -> directly into d_out at (b, t, :)
    k_gather_last<<<NBATCH, 256, 0, stream>>>(xB, lastB, t);
    k_gemm_bf16<<<gL, blk, 0, stream>>>(lastB, fcwB, fc_b,
                                        logitsBase + (size_t)t * NVOCAB, nullptr,
                                        NBATCH, NVOCAB, DMODEL, (long long)SEQ * NVOCAB, 0);
    k_argmax<<<NBATCH, 256, 0, stream>>>(logitsBase, t, tok, gen);
  }
}
